// TransformerLayerWithMoE_26843545600106
// MI455X (gfx1250) — matrix operand order repaired and checked by reading, not compiled, not run on hardware
//
#include <hip/hip_runtime.h>
#include <hip/hip_bf16.h>

#define D_    1024
#define H_    16
#define DH    64
#define E_    8
#define HID_  2048
#define B_    2
#define L_    2048
#define T_    (B_*L_)

#define MODE_QKV 0
#define MODE_WO  1
#define MODE_W1  2
#define MODE_W2  3

typedef __attribute__((ext_vector_type(16))) __bf16 v16bf;
typedef __attribute__((ext_vector_type(8)))  float  v8f;

union FragBF16 {
    v16bf  v;
    uint4  q[2];
    __bf16 e[16];
};

static __device__ __forceinline__ __bf16 f2bf(float f) { return (__bf16)f; }

// ---------------------------------------------------------------------------
// fp32 -> bf16 flat convert
// ---------------------------------------------------------------------------
__global__ __launch_bounds__(256)
void cvt_bf16_kernel(const float* __restrict__ in, __bf16* __restrict__ out, size_t n) {
    size_t i = (size_t)blockIdx.x * 256 + threadIdx.x;
    if (i < n) out[i] = f2bf(in[i]);
}

// per-expert transpose convert: out[e][c][r] = in[e][r][c]
__global__ __launch_bounds__(256)
void cvt_bf16_t_kernel(const float* __restrict__ in, __bf16* __restrict__ out, int R, int C) {
    size_t e = blockIdx.z;
    int idx = blockIdx.x * 256 + threadIdx.x;
    if (idx < R * C) {
        int r = idx / C, c = idx % C;
        out[e * (size_t)R * C + (size_t)c * R + r] =
            f2bf(in[e * (size_t)R * C + (size_t)r * C + c]);
    }
}

// ---------------------------------------------------------------------------
// LayerNorm (fp32 in, bf16 out), one row (D=1024) per block
// ---------------------------------------------------------------------------
__global__ __launch_bounds__(256)
void layernorm_kernel(const float* __restrict__ x, const float* __restrict__ g,
                      const float* __restrict__ b, __bf16* __restrict__ xbf) {
    __shared__ float red[8];
    int t = blockIdx.x, tid = threadIdx.x;
    const float* row = x + (size_t)t * D_;
    float s = 0.f;
    for (int i = tid; i < D_; i += 256) s += row[i];
    for (int m = 16; m >= 1; m >>= 1) s += __shfl_xor(s, m, 32);
    if ((tid & 31) == 0) red[tid >> 5] = s;
    __syncthreads();
    if (tid == 0) { float tot = 0; for (int i = 0; i < 8; i++) tot += red[i]; red[0] = tot / D_; }
    __syncthreads();
    float mu = red[0];
    __syncthreads();
    float v = 0.f;
    for (int i = tid; i < D_; i += 256) { float d = row[i] - mu; v += d * d; }
    for (int m = 16; m >= 1; m >>= 1) v += __shfl_xor(v, m, 32);
    if ((tid & 31) == 0) red[tid >> 5] = v;
    __syncthreads();
    if (tid == 0) { float tot = 0; for (int i = 0; i < 8; i++) tot += red[i]; red[0] = rsqrtf(tot / D_ + 1e-5f); }
    __syncthreads();
    float rstd = red[0];
    for (int i = tid; i < D_; i += 256)
        xbf[(size_t)t * D_ + i] = f2bf((row[i] - mu) * rstd * g[i] + b[i]);
}

// ---------------------------------------------------------------------------
// Tiled bf16 WMMA GEMM: C[M,N] = A[M,K] * B[N,K]^T (+ mode-specific epilogue)
// Block tile 128x128, BK=32, 8 waves in 4(M)x2(N); wave tile 32x64 = 8 WMMAs.
// Double-buffered LDS; fragment loads batched ahead of the WMMA group.
// ---------------------------------------------------------------------------
__global__ __launch_bounds__(256)
void gemm_bf16_kernel(const __bf16* __restrict__ A, const __bf16* __restrict__ Bw,
                      int M, int N, int K, int mode,
                      const float* __restrict__ bias,
                      const float* __restrict__ residual,
                      float* __restrict__ outf,
                      __bf16* __restrict__ qb, __bf16* __restrict__ kb, __bf16* __restrict__ vb,
                      __bf16* __restrict__ outb,
                      const int* __restrict__ cnt, const int* __restrict__ offs,
                      const int* __restrict__ btok, const float* __restrict__ bwt) {
    __shared__ __align__(16) __bf16 As[2][128 * 32];
    __shared__ __align__(16) __bf16 Bs[2][128 * 32];

    int tid = threadIdx.x, wave = tid >> 5, lane = tid & 31;
    int m0 = blockIdx.y * 128, n0 = blockIdx.x * 128;
    int e = blockIdx.z;

    int Meff = M;
    const __bf16* Bp = Bw;
    const float* biasp = bias;
    const int* gather = nullptr;
    int ob = 0, abase = 0;

    if (mode == MODE_W1) {
        Meff = cnt[e];
        Bp = Bw + (size_t)e * N * K;
        biasp = bias + (size_t)e * N;
        gather = btok + (size_t)e * T_;
        ob = offs[e];
        if (m0 >= Meff) return;
    } else if (mode == MODE_W2) {
        Meff = cnt[e];
        Bp = Bw + (size_t)e * N * K;
        biasp = bias + (size_t)e * N;
        ob = offs[e];
        abase = ob;
        if (m0 >= Meff) return;
    }

    // K-invariant source pointers (gather hoisted out of the K loop)
    int lr = tid >> 1, hseg = (tid & 1) * 16;
    const __bf16* asrc = nullptr;
    {
        int pos = m0 + lr;
        if (pos < Meff) {
            long arow = gather ? (long)gather[pos] : (long)(abase + pos);
            asrc = A + (size_t)arow * K + hseg;
        }
    }
    const __bf16* bsrc = nullptr;
    {
        int nn = n0 + lr;
        if (nn < N) bsrc = Bp + (size_t)nn * K + hseg;
    }

    uint4 ra0, ra1, rb0, rb1;
    auto gload = [&](int k0) {
        uint4 z = {0, 0, 0, 0};
        ra0 = ra1 = rb0 = rb1 = z;
        if (asrc) { const uint4* p = (const uint4*)(asrc + k0); ra0 = p[0]; ra1 = p[1]; }
        if (bsrc) { const uint4* p = (const uint4*)(bsrc + k0); rb0 = p[0]; rb1 = p[1]; }
    };
    auto sstore = [&](int buf) {
        uint4* da = (uint4*)(&As[buf][lr * 32 + hseg]);
        da[0] = ra0; da[1] = ra1;
        uint4* db = (uint4*)(&Bs[buf][lr * 32 + hseg]);
        db[0] = rb0; db[1] = rb1;
    };

    int wm = (wave & 3) * 32;   // wave row offset in block tile
    int wn = (wave >> 2) * 64;  // wave col offset in block tile

    v8f acc[2][4] = {};
    gload(0);
    sstore(0);
    int cur = 0;

    for (int k0 = 0; k0 < K; k0 += 32) {
        bool nxt = (k0 + 32) < K;
        if (nxt) gload(k0 + 32);       // global loads in flight during WMMAs
        __syncthreads();               // buffer `cur` ready

        FragBF16 af[2];
        for (int c = 0; c < 2; c++) {
            int row = wm + c * 16 + (lane & 15);
            int off = (lane >> 4) * 8;
            af[c].q[0] = *(const uint4*)(&As[cur][row * 32 + off]);
            af[c].q[1] = *(const uint4*)(&As[cur][row * 32 + off + 16]);
        }
        FragBF16 bf[4];
        for (int j = 0; j < 4; j++) {
            int rn = wn + j * 16 + (lane & 15);
            int off = (lane >> 4) * 8;
            bf[j].q[0] = *(const uint4*)(&Bs[cur][rn * 32 + off]);
            bf[j].q[1] = *(const uint4*)(&Bs[cur][rn * 32 + off + 16]);
        }
        for (int c = 0; c < 2; c++)
            for (int j = 0; j < 4; j++)
                acc[c][j] = __builtin_amdgcn_wmma_f32_16x16x32_bf16(
                    false, af[c].v, false, bf[j].v, (short)0, acc[c][j], false, false);

        if (nxt) sstore(cur ^ 1);      // fill the other buffer
        cur ^= 1;
    }

    // epilogue
    for (int c = 0; c < 2; c++) {
        for (int j = 0; j < 4; j++) {
            int gn = n0 + wn + j * 16 + (lane & 15);
            if (gn >= N) continue;
            for (int r = 0; r < 8; ++r) {
                int ml = ((lane >> 4) << 3) + r;
                int pos = m0 + wm + c * 16 + ml;
                if (pos >= Meff) continue;
                float v = acc[c][j][r];
                if (mode == MODE_QKV) {
                    float val = v + bias[gn];
                    int t = pos, bb = t / L_, l = t % L_;
                    int which = gn >> 10, di = gn & 1023;
                    int hh = di >> 6, dd = di & 63;
                    size_t dst = ((size_t)(bb * H_ + hh) * L_ + l) * DH + dd;
                    if (which == 0)      qb[dst] = f2bf(val * 0.125f); // 1/sqrt(64)
                    else if (which == 1) kb[dst] = f2bf(val);
                    else                 vb[dst] = f2bf(val);
                } else if (mode == MODE_WO) {
                    size_t o = (size_t)pos * D_ + gn;
                    outf[o] = residual[o] + v + bias[gn];
                } else if (mode == MODE_W1) {
                    float val = v + biasp[gn];
                    float ge = 0.5f * val * (1.f + erff(val * 0.70710678f));
                    outb[(size_t)(ob + pos) * N + gn] = f2bf(ge);
                } else { // MODE_W2
                    int tok = btok[(size_t)e * T_ + pos];
                    float wt = bwt[(size_t)e * T_ + pos];
                    atomicAdd(&outf[(size_t)tok * (size_t)N + gn], wt * (v + biasp[gn]));
                }
            }
        }
    }
}

// ---------------------------------------------------------------------------
// Flash attention: block = (b,h,q-tile of 128); 8 waves x 16 queries each.
// K staged row-major, V staged TRANSPOSED in LDS so the PV B-fragment is two
// aligned ds_load_b128 per lane instead of 16 scalar loads.
// ---------------------------------------------------------------------------
__global__ __launch_bounds__(256)
void attn_kernel(const __bf16* __restrict__ qb, const __bf16* __restrict__ kb,
                 const __bf16* __restrict__ vb, const unsigned char* __restrict__ mask,
                 __bf16* __restrict__ aob) {
    __shared__ __align__(16) __bf16 Kt[32 * 64];   // [key][dh]
    __shared__ __align__(16) __bf16 Vt[64 * 32];   // [dh][key] (transposed)
    __shared__ __align__(16) __bf16 Pt[8][16 * 32];
    __shared__ float Mk[32];

    int tid = threadIdx.x, wave = tid >> 5, lane = tid & 31;
    int b = blockIdx.z, h = blockIdx.y;
    int q0 = blockIdx.x * 128 + wave * 16;
    size_t base = ((size_t)(b * H_) + h) * L_;

    FragBF16 qa[2];
    {
        int row = q0 + (lane & 15);
        const __bf16* qrow = qb + (base + row) * DH;
        for (int c = 0; c < 2; c++) {
            int off = c * 32 + (lane >> 4) * 8;
            qa[c].q[0] = *(const uint4*)(qrow + off);
            qa[c].q[1] = *(const uint4*)(qrow + off + 16);
        }
    }

    float mrun[8], lrun[8];
    v8f o[4] = {};
    for (int r = 0; r < 8; r++) { mrun[r] = -3.0e38f; lrun[r] = 0.f; }

    for (int kt = 0; kt < L_; kt += 32) {
        __syncthreads();
        {   // cooperative tile load: key row = tid>>3, dh seg of 8
            int r = tid >> 3, seg = (tid & 7) * 8;
            *(uint4*)(Kt + r * 64 + seg) =
                *(const uint4*)(kb + (base + kt + r) * DH + seg);
            uint4 vv = *(const uint4*)(vb + (base + kt + r) * DH + seg);
            const __bf16* pv = (const __bf16*)&vv;
            for (int i = 0; i < 8; i++) Vt[(seg + i) * 32 + r] = pv[i];
            if (tid < 32) Mk[tid] = mask[(size_t)b * L_ + kt + tid] ? -1.0e9f : 0.f;
        }
        __syncthreads();

        // ---- scores: batch all 4 K-fragments, then 4 WMMAs ----
        FragBF16 kf[2][2];
        for (int jt = 0; jt < 2; jt++)
            for (int c = 0; c < 2; c++) {
                int rn = jt * 16 + (lane & 15);
                int off = c * 32 + (lane >> 4) * 8;
                kf[jt][c].q[0] = *(const uint4*)(Kt + rn * 64 + off);
                kf[jt][c].q[1] = *(const uint4*)(Kt + rn * 64 + off + 16);
            }
        v8f s[2] = {};
        s[0] = __builtin_amdgcn_wmma_f32_16x16x32_bf16(false, qa[0].v, false, kf[0][0].v, (short)0, s[0], false, false);
        s[0] = __builtin_amdgcn_wmma_f32_16x16x32_bf16(false, qa[1].v, false, kf[0][1].v, (short)0, s[0], false, false);
        s[1] = __builtin_amdgcn_wmma_f32_16x16x32_bf16(false, qa[0].v, false, kf[1][0].v, (short)0, s[1], false, false);
        s[1] = __builtin_amdgcn_wmma_f32_16x16x32_bf16(false, qa[1].v, false, kf[1][1].v, (short)0, s[1], false, false);
        for (int jt = 0; jt < 2; jt++) {
            float madd = Mk[jt * 16 + (lane & 15)];
            for (int r = 0; r < 8; r++) s[jt][r] += madd;
        }

        // ---- online softmax over these 32 keys ----
        float scl[8];
        for (int r = 0; r < 8; r++) {
            float v = fmaxf(s[0][r], s[1][r]);
            for (int m = 8; m >= 1; m >>= 1) v = fmaxf(v, __shfl_xor(v, m, 32));
            float mn = fmaxf(mrun[r], v);
            scl[r] = __expf(mrun[r] - mn);
            mrun[r] = mn;
            float p0 = __expf(s[0][r] - mn), p1 = __expf(s[1][r] - mn);
            s[0][r] = p0; s[1][r] = p1;
            float ssum = p0 + p1;
            for (int m = 8; m >= 1; m >>= 1) ssum += __shfl_xor(ssum, m, 32);
            lrun[r] = lrun[r] * scl[r] + ssum;
        }
        for (int nt = 0; nt < 4; nt++)
            for (int r = 0; r < 8; r++) o[nt][r] *= scl[r];

        // ---- P (C-layout) -> LDS -> A-layout fragment ----
        for (int jt = 0; jt < 2; jt++)
            for (int r = 0; r < 8; r++) {
                int ml = ((lane >> 4) << 3) + r;
                Pt[wave][ml * 32 + jt * 16 + (lane & 15)] = f2bf(s[jt][r]);
            }
        __syncthreads();

        FragBF16 pa;
        {
            int row = lane & 15, off = (lane >> 4) * 8;
            pa.q[0] = *(const uint4*)(&Pt[wave][row * 32 + off]);
            pa.q[1] = *(const uint4*)(&Pt[wave][row * 32 + off + 16]);
        }
        // ---- PV: batch all 4 V-fragments (vectorized via transposed Vt) ----
        FragBF16 vf[4];
        for (int nt = 0; nt < 4; nt++) {
            int n = nt * 16 + (lane & 15);
            int kb0 = (lane >> 4) * 8;
            vf[nt].q[0] = *(const uint4*)(Vt + n * 32 + kb0);
            vf[nt].q[1] = *(const uint4*)(Vt + n * 32 + kb0 + 16);
        }
        for (int nt = 0; nt < 4; nt++)
            o[nt] = __builtin_amdgcn_wmma_f32_16x16x32_bf16(
                false, pa.v, false, vf[nt].v, (short)0, o[nt], false, false);
    }

    for (int nt = 0; nt < 4; nt++)
        for (int r = 0; r < 8; r++) {
            int ml = ((lane >> 4) << 3) + r;
            int q = q0 + ml;
            int t = b * L_ + q;
            aob[(size_t)t * D_ + h * DH + nt * 16 + (lane & 15)] = f2bf(o[nt][r] / lrun[r]);
        }
}

// ---------------------------------------------------------------------------
// Router: 1 wave per token, logits over E=8, softmax, top-2, bucket build
// ---------------------------------------------------------------------------
__global__ __launch_bounds__(256)
void gate_kernel(const __bf16* __restrict__ xn, const float* __restrict__ Wg,
                 const float* __restrict__ bg, float* __restrict__ probs,
                 int* __restrict__ cnt, int* __restrict__ btok, float* __restrict__ bwt) {
    int wave = threadIdx.x >> 5, lane = threadIdx.x & 31;
    int t = blockIdx.x * 8 + wave;
    float acc[E_] = {};
    const __bf16* xr = xn + (size_t)t * D_;
    for (int d = lane; d < D_; d += 32) {
        float xv = (float)xr[d];
        const float* wr = Wg + (size_t)d * E_;
        for (int e = 0; e < E_; e++) acc[e] += xv * wr[e];
    }
    for (int e = 0; e < E_; e++)
        for (int m = 16; m >= 1; m >>= 1) acc[e] += __shfl_xor(acc[e], m, 32);
    if (lane == 0) {
        float mx = -3e38f;
        for (int e = 0; e < E_; e++) { acc[e] += bg[e]; mx = fmaxf(mx, acc[e]); }
        float sum = 0;
        for (int e = 0; e < E_; e++) { acc[e] = __expf(acc[e] - mx); sum += acc[e]; }
        float inv = 1.f / sum;
        int i0 = 0; float p0 = -1.f;
        for (int e = 0; e < E_; e++) {
            float p = acc[e] * inv;
            probs[(size_t)t * E_ + e] = p;
            if (p > p0) { p0 = p; i0 = e; }
        }
        int i1 = 0; float p1 = -1.f;
        for (int e = 0; e < E_; e++) {
            if (e == i0) continue;
            float p = acc[e] * inv;
            if (p > p1) { p1 = p; i1 = e; }
        }
        float ws = p0 + p1;
        int s0 = atomicAdd(&cnt[i0], 1);
        btok[i0 * T_ + s0] = t; bwt[i0 * T_ + s0] = p0 / ws;
        int s1 = atomicAdd(&cnt[i1], 1);
        btok[i1 * T_ + s1] = t; bwt[i1 * T_ + s1] = p1 / ws;
    }
}

__global__ void zero_int_kernel(int* p, int n) {
    int i = blockIdx.x * 256 + threadIdx.x;
    if (i < n) p[i] = 0;
}

__global__ void offs_kernel(const int* __restrict__ cnt, int* __restrict__ offs) {
    if (threadIdx.x == 0 && blockIdx.x == 0) {
        int o = 0;
        for (int e = 0; e < E_; e++) { offs[e] = o; o += cnt[e]; }
    }
}

__global__ __launch_bounds__(256)
void copy_f32_kernel(const float* __restrict__ a, float* __restrict__ o, size_t n) {
    size_t i = (size_t)blockIdx.x * 256 + threadIdx.x;
    if (i < n) o[i] = a[i];
}

__global__ __launch_bounds__(256)
void loss_kernel(const float* __restrict__ probs, const int* __restrict__ cnt,
                 float* __restrict__ loss_out) {
    __shared__ float me[E_];
    if (threadIdx.x < E_) me[threadIdx.x] = 0.f;
    __syncthreads();
    float acc[E_] = {};
    for (int t = threadIdx.x; t < T_; t += 256)
        for (int e = 0; e < E_; e++) acc[e] += probs[(size_t)t * E_ + e];
    for (int e = 0; e < E_; e++) {
        for (int m = 16; m >= 1; m >>= 1) acc[e] += __shfl_xor(acc[e], m, 32);
        if ((threadIdx.x & 31) == 0) atomicAdd(&me[e], acc[e]);
    }
    __syncthreads();
    if (threadIdx.x == 0) {
        float l = 0;
        for (int e = 0; e < E_; e++)
            l += (me[e] / (float)T_) * ((float)cnt[e] / (float)T_);
        loss_out[0] = (float)E_ * l;
    }
}

// ---------------------------------------------------------------------------
extern "C" void kernel_launch(void* const* d_in, const int* in_sizes, int n_in,
                              void* d_out, int out_size, void* d_ws, size_t ws_size,
                              hipStream_t stream) {
    const float* x     = (const float*)d_in[0];
    const unsigned char* mask = (const unsigned char*)d_in[1];
    const float* ln1_g = (const float*)d_in[2];
    const float* ln1_b = (const float*)d_in[3];
    const float* Wqkv  = (const float*)d_in[4];
    const float* bqkv  = (const float*)d_in[5];
    const float* Wo    = (const float*)d_in[6];
    const float* bo    = (const float*)d_in[7];
    const float* ln2_g = (const float*)d_in[8];
    const float* ln2_b = (const float*)d_in[9];
    const float* Wg    = (const float*)d_in[10];
    const float* bg    = (const float*)d_in[11];
    const float* W1    = (const float*)d_in[12];
    const float* b1    = (const float*)d_in[13];
    const float* W2    = (const float*)d_in[14];
    const float* b2    = (const float*)d_in[15];
    float* out = (float*)d_out;

    char* ws = (char*)d_ws;
    size_t cur = 0;
    auto alloc = [&](size_t bytes) -> char* {
        char* p = ws + cur;
        cur += (bytes + 255) & ~(size_t)255;
        return p;
    };

    __bf16* xn1    = (__bf16*)alloc((size_t)T_ * D_ * 2);
    __bf16* wqkv_b = (__bf16*)alloc((size_t)3 * D_ * D_ * 2);
    __bf16* wo_b   = (__bf16*)alloc((size_t)D_ * D_ * 2);
    __bf16* w1t_b  = (__bf16*)alloc((size_t)E_ * D_ * HID_ * 2);
    __bf16* w2t_b  = (__bf16*)alloc((size_t)E_ * HID_ * D_ * 2);
    __bf16* qbuf   = (__bf16*)alloc((size_t)T_ * D_ * 2);
    __bf16* kbuf   = (__bf16*)alloc((size_t)T_ * D_ * 2);
    __bf16* vbuf   = (__bf16*)alloc((size_t)T_ * D_ * 2);
    __bf16* aob    = (__bf16*)alloc((size_t)T_ * D_ * 2);
    float*  x1     = (float*)alloc((size_t)T_ * D_ * 4);
    __bf16* xn2    = (__bf16*)alloc((size_t)T_ * D_ * 2);
    float*  probs  = (float*)alloc((size_t)T_ * E_ * 4);
    int*    cnt    = (int*)alloc(E_ * 4);
    int*    offs   = (int*)alloc(E_ * 4);
    int*    btok   = (int*)alloc((size_t)E_ * T_ * 4);
    float*  bwt    = (float*)alloc((size_t)E_ * T_ * 4);
    __bf16* hbuf   = (__bf16*)alloc((size_t)2 * T_ * HID_ * 2);
    (void)ws_size; (void)n_in; (void)in_sizes; (void)out_size;

    // 1) weight converts
    {
        size_t n = (size_t)3 * D_ * D_;
        cvt_bf16_kernel<<<dim3((unsigned)((n + 255) / 256)), 256, 0, stream>>>(Wqkv, wqkv_b, n);
        n = (size_t)D_ * D_;
        cvt_bf16_kernel<<<dim3((unsigned)((n + 255) / 256)), 256, 0, stream>>>(Wo, wo_b, n);
        int rc = D_ * HID_;
        cvt_bf16_t_kernel<<<dim3((rc + 255) / 256, 1, E_), 256, 0, stream>>>(W1, w1t_b, D_, HID_);  // -> [e][hid][d]
        cvt_bf16_t_kernel<<<dim3((rc + 255) / 256, 1, E_), 256, 0, stream>>>(W2, w2t_b, HID_, D_);  // -> [e][d][hid]
    }

    // 2) LN1
    layernorm_kernel<<<T_, 256, 0, stream>>>(x, ln1_g, ln1_b, xn1);

    // 3) QKV GEMM (M=4096, N=3072, K=1024)
    gemm_bf16_kernel<<<dim3(3 * D_ / 128, T_ / 128, 1), 256, 0, stream>>>(
        xn1, wqkv_b, T_, 3 * D_, D_, MODE_QKV, bqkv, nullptr, nullptr,
        qbuf, kbuf, vbuf, nullptr, cnt, offs, btok, bwt);

    // 4) attention
    attn_kernel<<<dim3(L_ / 128, H_, B_), 256, 0, stream>>>(qbuf, kbuf, vbuf, mask, aob);

    // 5) Wo GEMM + residual -> x1
    gemm_bf16_kernel<<<dim3(D_ / 128, T_ / 128, 1), 256, 0, stream>>>(
        aob, wo_b, T_, D_, D_, MODE_WO, bo, x, x1,
        nullptr, nullptr, nullptr, nullptr, cnt, offs, btok, bwt);

    // 6) LN2
    layernorm_kernel<<<T_, 256, 0, stream>>>(x1, ln2_g, ln2_b, xn2);

    // 7) routing
    zero_int_kernel<<<1, 256, 0, stream>>>(cnt, E_);
    gate_kernel<<<T_ / 8, 256, 0, stream>>>(xn2, Wg, bg, probs, cnt, btok, bwt);
    offs_kernel<<<1, 32, 0, stream>>>(cnt, offs);

    // 8) expert FFN up (gathered, top-2 only): h = gelu(x @ W1[e] + b1[e])
    gemm_bf16_kernel<<<dim3(HID_ / 128, T_ / 128, E_), 256, 0, stream>>>(
        xn2, w1t_b, T_, HID_, D_, MODE_W1, b1, nullptr, nullptr,
        nullptr, nullptr, nullptr, hbuf, cnt, offs, btok, bwt);

    // 9) out = x1 (residual), then scatter-add weighted expert outputs
    copy_f32_kernel<<<(unsigned)(((size_t)T_ * D_ + 255) / 256), 256, 0, stream>>>(
        x1, out, (size_t)T_ * D_);
    gemm_bf16_kernel<<<dim3(D_ / 128, T_ / 128, E_), 256, 0, stream>>>(
        hbuf, w2t_b, T_, D_, HID_, MODE_W2, b2, nullptr, out,
        nullptr, nullptr, nullptr, nullptr, cnt, offs, btok, bwt);

    // 10) aux loss
    loss_kernel<<<1, 256, 0, stream>>>(probs, cnt, out + (size_t)T_ * D_);
}
